// PubMedGAT_56796647522839
// MI455X (gfx1250) — matrix operand order repaired and checked by reading, not compiled, not run on hardware
//
#include <hip/hip_runtime.h>

typedef __attribute__((ext_vector_type(16))) __bf16 v16bf;
typedef __attribute__((ext_vector_type(8)))  float  v8f;

#define N_NODES 100000
#define N_EDGES 1600000
#define F_IN    500
#define KPAD1   512
#define C1      128   // heads * hidden
#define HEADS   4
#define C2      3

__device__ __forceinline__ unsigned short f2bf(float f) {
  unsigned u = __builtin_bit_cast(unsigned, f);
  u += 0x7FFFu + ((u >> 16) & 1u);   // round-to-nearest-even
  return (unsigned short)(u >> 16);
}

__device__ __forceinline__ float elu1(float x) {
  return x > 0.f ? x : (__expf(x) - 1.f);
}

// ---------------------------------------------------------------------------
// Pack W1 [500,128] f32 -> bf16 fragments: [(kt*8+nt)*32 + lane]*16 + i
// B-fragment layout (16-bit, 32x16): lane L holds col n=L%16, k = (i%8) + 8*(L/16) + 16*(i/8)
// ---------------------------------------------------------------------------
__global__ void pack_w1(const float* __restrict__ W1, unsigned short* __restrict__ W1p) {
  int tid = blockIdx.x * blockDim.x + threadIdx.x;   // 16*8*32*16 = 65536
  int i    = tid & 15;
  int lane = (tid >> 4) & 31;
  int nt   = (tid >> 9) & 7;
  int kt   = tid >> 12;
  int k   = kt * 32 + (i & 7) + ((lane >> 4) << 3) + ((i >> 3) << 4);
  int col = nt * 16 + (lane & 15);
  int kc  = (k < F_IN) ? k : 0;
  float v = W1[kc * C1 + col];
  W1p[tid] = f2bf((k < F_IN) ? v : 0.f);
}

// Pack W2 [128,3] f32 -> bf16 fragments, single N-tile padded to 16 cols.
__global__ void pack_w2(const float* __restrict__ W2, unsigned short* __restrict__ W2p) {
  int tid = blockIdx.x * blockDim.x + threadIdx.x;   // 4*32*16 = 2048
  int i    = tid & 15;
  int lane = (tid >> 4) & 31;
  int kt   = tid >> 9;
  int k   = kt * 32 + (i & 7) + ((lane >> 4) << 3) + ((i >> 3) << 4);        // < 128 always
  int col = lane & 15;
  int cc  = (col < C2) ? col : 0;
  float v = W2[k * C2 + cc];
  W2p[tid] = f2bf((col < C2) ? v : 0.f);
}

// ---------------------------------------------------------------------------
// GEMM1: z1[100000,128] = x[100000,500] @ W1.
// 64 rows/block (4 sub-tiles of 16), 8 waves each owning one 16-col tile.
// Per k-tile: each wave loads its B fragment ONCE and issues 4 WMMAs.
// A staged into LDS in fragment order, as PACKED bf16 pairs (float2 load,
// one ds_store_b32 per pair). Guards are branch-free (clamped addresses +
// cndmask) so staging loads batch into clauses with a single wait.
// ---------------------------------------------------------------------------
__global__ __launch_bounds__(256) void gemm1(const float* __restrict__ x,
                                             const unsigned short* __restrict__ W1p,
                                             float* __restrict__ z1) {
  __shared__ alignas(32) unsigned As32[4 * 256];   // 64 rows x 32 k (bf16, packed)
  const int t = threadIdx.x;
  const int wave = t >> 5, lane = t & 31;
  const int row0 = blockIdx.x * 64;
  v8f c[4] = {v8f{}, v8f{}, v8f{}, v8f{}};
  // prefetch row/line this thread covers (4 threads per row, 32B apart)
  const int pfrow_u = row0 + (t >> 2);
  const int pfrow   = (pfrow_u < N_NODES) ? pfrow_u : (N_NODES - 1);
  const int pfoff   = (t & 3) * 8;

  for (int kt = 0; kt < KPAD1 / 32; ++kt) {
    const int kb = kt * 32;
    if (kt + 1 < KPAD1 / 32)   // uniform branch: prefetch next k-tile of x
      __builtin_prefetch(x + (size_t)pfrow * F_IN + kb + 32 + pfoff, 0, 1);
    // stage 64x32 A tile into fragment-ordered LDS (bf16 pair per store)
#pragma unroll
    for (int q = 0; q < 4; ++q) {
      int e = t + q * 256;           // 1024 pairs: (m in 0..63, p in 0..15)
      int m = e >> 4;
      int p = e & 15;
      int row = row0 + m;
      int k   = kb + 2 * p;
      bool ok = (row < N_NODES) && (k + 1 < F_IN);   // F_IN even: pair all-or-none
      int rowc = (row < N_NODES) ? row : (N_NODES - 1);
      int kc   = ok ? k : 0;
      const float2 xv = *reinterpret_cast<const float2*>(x + (size_t)rowc * F_IN + kc);
      float v0 = ok ? xv.x : 0.f;
      float v1 = ok ? xv.y : 0.f;
      int st = m >> 4, ml = m & 15;
      int fl = ml + (((p & 7) >= 4) ? 16 : 0);
      int j  = (p & 3) + ((p >> 3) << 2);
      As32[st * 256 + fl * 8 + j] = (unsigned)f2bf(v0) | ((unsigned)f2bf(v1) << 16);
    }
    __syncthreads();
    v16bf b = *reinterpret_cast<const v16bf*>(&W1p[((kt * 8 + wave) * 32 + lane) * 16]);
#pragma unroll
    for (int st = 0; st < 4; ++st) {
      v16bf a = *reinterpret_cast<const v16bf*>(&As32[st * 256 + lane * 8]);
      c[st] = __builtin_amdgcn_wmma_f32_16x16x32_bf16(false, a, false, b, (short)0,
                                                      c[st], false, false);
    }
    __syncthreads();
  }
  const int col   = wave * 16 + (lane & 15);
  const int rbase = (lane < 16) ? 0 : 8;
#pragma unroll
  for (int st = 0; st < 4; ++st) {
#pragma unroll
    for (int r = 0; r < 8; ++r) {
      int row = row0 + st * 16 + rbase + r;
      if (row < N_NODES) z1[(size_t)row * C1 + col] = c[st][r];
    }
  }
}

// ---------------------------------------------------------------------------
// Attention projections layer 1: el1/er1[n,h] = sum_d z1[n,h,d] * a[h,d]
// One wave per node; lane owns 4 columns; shuffle-reduce within 8-lane head.
// ---------------------------------------------------------------------------
__global__ __launch_bounds__(256) void att1(const float* __restrict__ z1,
                                            const float* __restrict__ al1,
                                            const float* __restrict__ ar1,
                                            float* __restrict__ el1,
                                            float* __restrict__ er1) {
  int n = (blockIdx.x * 256 + threadIdx.x) >> 5;
  if (n >= N_NODES) return;
  int lane = threadIdx.x & 31;
  int col  = lane * 4;
  const float4 z = *reinterpret_cast<const float4*>(z1 + (size_t)n * C1 + col);
  const float4 a = *reinterpret_cast<const float4*>(al1 + col);
  const float4 r = *reinterpret_cast<const float4*>(ar1 + col);
  float vl = z.x * a.x + z.y * a.y + z.z * a.z + z.w * a.w;
  float vr = z.x * r.x + z.y * r.y + z.z * r.z + z.w * r.w;
#pragma unroll
  for (int k = 1; k < 8; k <<= 1) {
    vl += __shfl_xor(vl, k, 32);
    vr += __shfl_xor(vr, k, 32);
  }
  if ((lane & 7) == 0) {
    int hh = lane >> 3;
    el1[n * HEADS + hh] = vl;
    er1[n * HEADS + hh] = vr;
  }
}

// CSR rowptr over sorted dst (lower_bound per node). rowptr[N] == E.
__global__ void build_rowptr(const int* __restrict__ dst, int* __restrict__ rowptr) {
  int n = blockIdx.x * blockDim.x + threadIdx.x;
  if (n > N_NODES) return;
  int lo = 0, hi = N_EDGES;
  while (lo < hi) {
    int mid = (lo + hi) >> 1;
    if (dst[mid] < n) lo = mid + 1; else hi = mid;
  }
  rowptr[n] = lo;
}

// ---------------------------------------------------------------------------
// Edge pass 1: wave per dst node. Segmented (dst sorted) softmax + weighted
// aggregation with NO atomics: pass1 max, pass2 exp-sum + z1[src] accumulate,
// then divide, add bias, ELU -> h[100000,128].
// ---------------------------------------------------------------------------
__global__ __launch_bounds__(256) void edge1(const int* __restrict__ src,
                                             const int* __restrict__ rowptr,
                                             const float* __restrict__ z1,
                                             const float* __restrict__ el1,
                                             const float* __restrict__ er1,
                                             const float* __restrict__ b1,
                                             float* __restrict__ hout) {
  int n = (blockIdx.x * 256 + threadIdx.x) >> 5;
  if (n >= N_NODES) return;
  int lane = threadIdx.x & 31;
  int hh   = lane >> 3;          // head of this lane's 4 columns
  int col  = lane * 4;
  int beg = rowptr[n], end = rowptr[n + 1];
  float ern = er1[n * HEADS + hh];

  float m = -1e30f;
  for (int e = beg; e < end; ++e) {
    int s = src[e];
    float v = el1[s * HEADS + hh] + ern;
    v = (v > 0.f) ? v : 0.2f * v;          // leaky_relu
    m = fmaxf(m, v);
  }
  float denom = 0.f;
  float4 acc = {0.f, 0.f, 0.f, 0.f};
  for (int e = beg; e < end; ++e) {
    int s = src[e];
    float v = el1[s * HEADS + hh] + ern;
    v = (v > 0.f) ? v : 0.2f * v;
    float p = __expf(v - m);
    denom += p;
    const float4 z = *reinterpret_cast<const float4*>(z1 + (size_t)s * C1 + col);
    acc.x += p * z.x; acc.y += p * z.y; acc.z += p * z.z; acc.w += p * z.w;
  }
  float inv = 1.f / fmaxf(denom, 1e-9f);
  const float4 bb = *reinterpret_cast<const float4*>(b1 + col);
  float* hp = hout + (size_t)n * C1 + col;
  hp[0] = elu1(acc.x * inv + bb.x);
  hp[1] = elu1(acc.y * inv + bb.y);
  hp[2] = elu1(acc.z * inv + bb.z);
  hp[3] = elu1(acc.w * inv + bb.w);
}

// ---------------------------------------------------------------------------
// GEMM2: z2[100000,3] = h[100000,128] @ W2 (N padded to 16). 8 waves/block,
// each wave owns 16 rows; K loop of 4 bf16 WMMA steps. Branch-free staging.
// ---------------------------------------------------------------------------
__global__ __launch_bounds__(256) void gemm2(const float* __restrict__ hin,
                                             const unsigned short* __restrict__ W2p,
                                             float* __restrict__ z2) {
  __shared__ alignas(32) unsigned short As[8 * 512];
  const int t = threadIdx.x;
  const int wave = t >> 5, lane = t & 31;
  const int row0 = blockIdx.x * 128 + wave * 16;
  unsigned short* Aw = &As[wave * 512];
  v8f c = {};
  for (int kt = 0; kt < 4; ++kt) {
    const int kb = kt * 32;
    const int kk = lane;
    const int fl = ((kk & 15) >= 8) ? 16 : 0;
    const int j  = ((kk & 7) >> 1) + ((kk >> 4) << 2);
    const int fi = 2 * j + (kk & 1);
#pragma unroll
    for (int s = 0; s < 16; ++s) {
      int row  = row0 + s;
      int rowc = (row < N_NODES) ? row : (N_NODES - 1);
      float v  = hin[(size_t)rowc * C1 + kb + kk];
      v = (row < N_NODES) ? v : 0.f;
      Aw[(s + fl) * 16 + fi] = f2bf(v);
    }
    __syncthreads();
    v16bf a = *reinterpret_cast<const v16bf*>(&Aw[lane * 16]);
    v16bf b = *reinterpret_cast<const v16bf*>(&W2p[(kt * 32 + lane) * 16]);
    c = __builtin_amdgcn_wmma_f32_16x16x32_bf16(false, a, false, b, (short)0, c, false, false);
    __syncthreads();
  }
  const int col = lane & 15;
  if (col < C2) {
    const int rbase = (lane < 16) ? 0 : 8;
#pragma unroll
    for (int r = 0; r < 8; ++r) {
      int row = row0 + rbase + r;
      if (row < N_NODES) z2[row * C2 + col] = c[r];
    }
  }
}

// Attention projections layer 2 (single head, D=3): thread per node.
__global__ void att2(const float* __restrict__ z2,
                     const float* __restrict__ al2, const float* __restrict__ ar2,
                     float* __restrict__ el2, float* __restrict__ er2) {
  int n = blockIdx.x * blockDim.x + threadIdx.x;
  if (n >= N_NODES) return;
  float a = z2[n * 3 + 0], b = z2[n * 3 + 1], c = z2[n * 3 + 2];
  el2[n] = a * al2[0] + b * al2[1] + c * al2[2];
  er2[n] = a * ar2[0] + b * ar2[1] + c * ar2[2];
}

// Edge pass 2: thread per dst node (D=3), writes final output [N,1,3].
__global__ void edge2(const int* __restrict__ src, const int* __restrict__ rowptr,
                      const float* __restrict__ z2,
                      const float* __restrict__ el2, const float* __restrict__ er2,
                      const float* __restrict__ b2, float* __restrict__ out) {
  int n = blockIdx.x * blockDim.x + threadIdx.x;
  if (n >= N_NODES) return;
  int beg = rowptr[n], end = rowptr[n + 1];
  float ern = er2[n];
  float m = -1e30f;
  for (int e = beg; e < end; ++e) {
    float v = el2[src[e]] + ern;
    v = (v > 0.f) ? v : 0.2f * v;
    m = fmaxf(m, v);
  }
  float denom = 0.f, a0 = 0.f, a1 = 0.f, a2 = 0.f;
  for (int e = beg; e < end; ++e) {
    int s = src[e];
    float v = el2[s] + ern;
    v = (v > 0.f) ? v : 0.2f * v;
    float p = __expf(v - m);
    denom += p;
    a0 += p * z2[s * 3 + 0];
    a1 += p * z2[s * 3 + 1];
    a2 += p * z2[s * 3 + 2];
  }
  float inv = 1.f / fmaxf(denom, 1e-9f);
  out[n * 3 + 0] = a0 * inv + b2[0];
  out[n * 3 + 1] = a1 * inv + b2[1];
  out[n * 3 + 2] = a2 * inv + b2[2];
}

// ---------------------------------------------------------------------------
extern "C" void kernel_launch(void* const* d_in, const int* in_sizes, int n_in,
                              void* d_out, int out_size, void* d_ws, size_t ws_size,
                              hipStream_t stream) {
  (void)in_sizes; (void)n_in; (void)out_size; (void)ws_size;
  const float* x   = (const float*)d_in[0];
  const int*   src = (const int*)d_in[1];
  const int*   dst = (const int*)d_in[2];
  const float* W1  = (const float*)d_in[3];
  const float* al1 = (const float*)d_in[4];
  const float* ar1 = (const float*)d_in[5];
  const float* b1  = (const float*)d_in[6];
  const float* W2  = (const float*)d_in[7];
  const float* al2 = (const float*)d_in[8];
  const float* ar2 = (const float*)d_in[9];
  const float* b2  = (const float*)d_in[10];
  float* out = (float*)d_out;

  char* ws = (char*)d_ws;
  size_t off = 0;
  auto alloc = [&](size_t bytes) -> void* {
    off = (off + 255) & ~(size_t)255;
    void* p = ws + off;
    off += bytes;
    return p;
  };
  unsigned short* W1p = (unsigned short*)alloc(16 * 8 * 32 * 16 * sizeof(unsigned short));
  unsigned short* W2p = (unsigned short*)alloc(4 * 32 * 16 * sizeof(unsigned short));
  float* z1   = (float*)alloc((size_t)N_NODES * C1 * sizeof(float));
  float* el1  = (float*)alloc((size_t)N_NODES * HEADS * sizeof(float));
  float* er1  = (float*)alloc((size_t)N_NODES * HEADS * sizeof(float));
  int*   rp   = (int*)alloc((size_t)(N_NODES + 1) * sizeof(int));
  float* hbuf = (float*)alloc((size_t)N_NODES * C1 * sizeof(float));
  float* z2   = (float*)alloc((size_t)N_NODES * C2 * sizeof(float));
  float* el2  = (float*)alloc((size_t)N_NODES * sizeof(float));
  float* er2  = (float*)alloc((size_t)N_NODES * sizeof(float));

  pack_w1<<<256, 256, 0, stream>>>(W1, W1p);
  pack_w2<<<8, 256, 0, stream>>>(W2, W2p);
  gemm1<<<(N_NODES + 63) / 64, 256, 0, stream>>>(x, W1p, z1);
  att1<<<(N_NODES * 32 + 255) / 256, 256, 0, stream>>>(z1, al1, ar1, el1, er1);
  build_rowptr<<<(N_NODES + 1 + 255) / 256, 256, 0, stream>>>(dst, rp);
  edge1<<<(N_NODES * 32 + 255) / 256, 256, 0, stream>>>(src, rp, z1, el1, er1, b1, hbuf);
  gemm2<<<(N_NODES + 127) / 128, 256, 0, stream>>>(hbuf, W2p, z2);
  att2<<<(N_NODES + 255) / 256, 256, 0, stream>>>(z2, al2, ar2, el2, er2);
  edge2<<<(N_NODES + 255) / 256, 256, 0, stream>>>(src, rp, z2, el2, er2, b2, out);
}
